// CustomLayer_5282809774772
// MI455X (gfx1250) — matrix operand order repaired and checked by reading, not compiled, not run on hardware
//
#include <hip/hip_runtime.h>

#define BATCH 8192
#define IN_F  1024
#define OUT_F 1024

typedef __attribute__((ext_vector_type(16))) _Float16 v16h;
typedef __attribute__((ext_vector_type(8)))  _Float16 v8h;
typedef __attribute__((ext_vector_type(4)))  _Float16 v4h;
typedef __attribute__((ext_vector_type(8)))  float    v8f;

// ---- monotonic uint encoding of float for atomic min/max over signed data ----
__device__ __forceinline__ unsigned fkey(float f) {
  unsigned u = __float_as_uint(f);
  return (u & 0x80000000u) ? ~u : (u | 0x80000000u);
}
__device__ __forceinline__ float funkey(unsigned k) {
  return (k & 0x80000000u) ? __uint_as_float(k ^ 0x80000000u)
                           : __uint_as_float(~k);
}

// ---------------------------------------------------------------- init ws ----
__global__ void xbar_init_kernel(unsigned* keys, float* sums) {
  int t = blockIdx.x * 256 + threadIdx.x;
  if (t == 0) { keys[0] = 0xFFFFFFFFu; keys[1] = 0u; }  // minKey, maxKey
  if (t < 2 * IN_F) sums[t] = 0.0f;                     // geff_sum | gq_sum
}

// ------------------------------------------------- global min/max of weight --
__global__ __launch_bounds__(256)
void xbar_minmax_kernel(const float* __restrict__ w, unsigned* keys) {
  const int NW = IN_F * OUT_F;
  float mn = 3.4e38f, mx = -3.4e38f;
  for (int i = blockIdx.x * blockDim.x + threadIdx.x; i < NW;
       i += gridDim.x * blockDim.x) {
    float v = w[i];
    mn = fminf(mn, v);
    mx = fmaxf(mx, v);
  }
#pragma unroll
  for (int off = 16; off; off >>= 1) {
    mn = fminf(mn, __shfl_xor(mn, off, 32));
    mx = fmaxf(mx, __shfl_xor(mx, off, 32));
  }
  __shared__ float smn[8], smx[8];
  int lane = threadIdx.x & 31, wv = threadIdx.x >> 5;
  if (lane == 0) { smn[wv] = mn; smx[wv] = mx; }
  __syncthreads();
  if (threadIdx.x == 0) {
    for (int i = 1; i < 8; i++) { mn = fminf(mn, smn[i]); mx = fmaxf(mx, smx[i]); }
    atomicMin(&keys[0], fkey(mn));
    atomicMax(&keys[1], fkey(mx));
  }
}

// ---- build W_eff (f16, [out][in] == [N][K]) + column sums of G_eff / G_q ----
__global__ __launch_bounds__(256)
void xbar_prep_kernel(const float* __restrict__ weight, const unsigned* keys,
                      float* __restrict__ geff_sum, float* __restrict__ gq_sum,
                      _Float16* __restrict__ Weff) {
  const float Wmin = funkey(keys[0]);
  const float Wmax = funkey(keys[1]);
  const float Ghrs = 1.0f / 1000000.0f, Glrs = 1.0f / 1000.0f;
  const float a = (Glrs - Ghrs) / (Wmax - Wmin);
  const float b = Ghrs - a * Wmin;
  const float inv_a = 1.0f / a;
  const float stepq = (Glrs - Ghrs) / 15.0f;
  const float inv_stepq = 15.0f / (Glrs - Ghrs);

  const int t = threadIdx.x;
  const int j0 = blockIdx.x * 16;  // 64 blocks cover OUT_F rows of weight
  float accE[4] = {0.f, 0.f, 0.f, 0.f};
  float accQ[4] = {0.f, 0.f, 0.f, 0.f};

  for (int jj = 0; jj < 16; jj++) {
    int j = j0 + jj;  // output column index
    float rj = 2.0f * (float)(j + 1);  // PARASITIC_R * (j+1)
#pragma unroll
    for (int c = 0; c < 4; c++) {
      int i = c * 256 + t;  // input row index
      float w = weight[(size_t)j * IN_F + i];
      float Gm = a * w + b;                                   // weight_mapping
      float Gq = rintf((Gm - Ghrs) * inv_stepq) * stepq + Ghrs;  // quantization
      float rs = rj + 2.0f * (float)(IN_F - i);               // r_series(i,j)
      float Geff = Gq / (1.0f + Gq * rs);                     // 1/(1/Gq + rs)
      accE[c] += Geff;
      accQ[c] += Gq;
      // fold output_unmapping into B-matrix: x @ Weff == (x@Geff - b*xsum)/a
      Weff[(size_t)j * IN_F + i] = (_Float16)((Geff - b) * inv_a);
    }
  }
#pragma unroll
  for (int c = 0; c < 4; c++) {
    atomicAdd(&geff_sum[c * 256 + t], accE[c]);
    atomicAdd(&gq_sum[c * 256 + t], accQ[c]);
  }
}

// ---- correction[b] = (x[b]·Σ_j Geff)/(x[b]·Σ_j Gq)  (means cancel) ---------
__global__ __launch_bounds__(256)
void xbar_corr_kernel(const float* __restrict__ x,
                      const float* __restrict__ geff_sum,
                      const float* __restrict__ gq_sum,
                      float* __restrict__ corrv) {
  const int lane = threadIdx.x & 31;
  const int m = blockIdx.x * 8 + (threadIdx.x >> 5);
  const float* xr = x + (size_t)m * IN_F;
  float de = 0.f, dq = 0.f;
  for (int k = lane; k < IN_F; k += 32) {
    float xv = xr[k];
    de += xv * geff_sum[k];
    dq += xv * gq_sum[k];
  }
#pragma unroll
  for (int off = 16; off; off >>= 1) {
    de += __shfl_down(de, off, 32);
    dq += __shfl_down(dq, off, 32);
  }
  if (lane == 0) corrv[m] = de / dq;
}

// ---- main GEMM: out = x @ Weff + bias ⊗ corr, f16 WMMA / f32 accumulate ----
// block tile 128x128x32, 8 waves as 4(M) x 2(N), wave tile 32x64 (2x4 WMMA)
#define SA 40  // padded LDS row stride in halfs (80B: 16B aligned, bank-spread)
__global__ __launch_bounds__(256)
void xbar_gemm_kernel(const float* __restrict__ x,
                      const _Float16* __restrict__ Weff,
                      const float* __restrict__ bias,
                      const float* __restrict__ corrv,
                      float* __restrict__ out) {
  __shared__ __align__(16) _Float16 As[128 * SA];
  __shared__ __align__(16) _Float16 Bs[128 * SA];

  const int tid  = threadIdx.x;
  const int lane = tid & 31;
  const int l16  = lane & 15;
  const int hsel = lane >> 4;
  const int wv   = tid >> 5;
  const int waveM = wv & 3;   // 0..3
  const int waveN = wv >> 2;  // 0..1
  const int mBlock = blockIdx.y * 128;
  const int nBlock = blockIdx.x * 128;

  // staging roles
  const int aRow  = tid >> 3;  // 0..31 (+32*r)
  const int aQuad = tid & 7;   // float4 slot within 32-wide k chunk
  const int bRow  = tid >> 2;  // 0..63 (+64*r)
  const int bSeg  = tid & 3;   // v8h slot within 32-wide k chunk

  float4 aReg[4];
  v8h    bReg[2];
  v8f acc[2][4] = {};

  // prefetch k-block 0 into registers
#pragma unroll
  for (int r = 0; r < 4; r++)
    aReg[r] = *(const float4*)(x + (size_t)(mBlock + aRow + r * 32) * IN_F + aQuad * 4);
#pragma unroll
  for (int r = 0; r < 2; r++)
    bReg[r] = *(const v8h*)(Weff + (size_t)(nBlock + bRow + r * 64) * IN_F + bSeg * 8);

  for (int kb = 0; kb < IN_F; kb += 32) {
    // commit prefetched tile to LDS (A converted f32 -> f16 here)
#pragma unroll
    for (int r = 0; r < 4; r++) {
      v4h h;
      h[0] = (_Float16)aReg[r].x; h[1] = (_Float16)aReg[r].y;
      h[2] = (_Float16)aReg[r].z; h[3] = (_Float16)aReg[r].w;
      *(v4h*)(&As[(aRow + r * 32) * SA + aQuad * 4]) = h;
    }
#pragma unroll
    for (int r = 0; r < 2; r++)
      *(v8h*)(&Bs[(bRow + r * 64) * SA + bSeg * 8]) = bReg[r];
    __syncthreads();

    // prefetch next k-block while WMMAs run
    const int kn = kb + 32;
    if (kn < IN_F) {
#pragma unroll
      for (int r = 0; r < 4; r++)
        aReg[r] = *(const float4*)(x + (size_t)(mBlock + aRow + r * 32) * IN_F + kn + aQuad * 4);
#pragma unroll
      for (int r = 0; r < 2; r++)
        bReg[r] = *(const v8h*)(Weff + (size_t)(nBlock + bRow + r * 64) * IN_F + kn + bSeg * 8);
    }

    // B fragments: lanes<16 K={0..7,16..23}, lanes>=16 K={8..15,24..31}
    v16h bf[4];
#pragma unroll
    for (int nf = 0; nf < 4; nf++) {
      const _Float16* p = &Bs[(waveN * 64 + nf * 16 + l16) * SA + hsel * 8];
      v8h lo = *(const v8h*)p;
      v8h hi = *(const v8h*)(p + 16);
      v16h tv;
#pragma unroll
      for (int i = 0; i < 8; i++) { tv[i] = lo[i]; tv[i + 8] = hi[i]; }
      bf[nf] = tv;
    }
    // A fragments: lanes<16 K={0..7,16..23}, lanes>=16 K={8..15,24..31}
#pragma unroll
    for (int mf = 0; mf < 2; mf++) {
      const _Float16* p = &As[(waveM * 32 + mf * 16 + l16) * SA + hsel * 8];
      v8h lo = *(const v8h*)p;
      v8h hi = *(const v8h*)(p + 16);
      v16h af;
#pragma unroll
      for (int i = 0; i < 8; i++) { af[i] = lo[i]; af[i + 8] = hi[i]; }
#pragma unroll
      for (int nf = 0; nf < 4; nf++) {
        acc[mf][nf] = __builtin_amdgcn_wmma_f32_16x16x32_f16(
            false, af, false, bf[nf], (short)0, acc[mf][nf], false, false);
      }
    }
    __syncthreads();
  }

  // epilogue: C layout — vgpr v, lane<16 -> M=v, lane>=16 -> M=v+8; N=l16
#pragma unroll
  for (int mf = 0; mf < 2; mf++) {
#pragma unroll
    for (int nf = 0; nf < 4; nf++) {
      int n = nBlock + waveN * 64 + nf * 16 + l16;
      float bn = bias[n];
#pragma unroll
      for (int v = 0; v < 8; v++) {
        int m = mBlock + waveM * 32 + mf * 16 + hsel * 8 + v;
        out[(size_t)m * OUT_F + n] = acc[mf][nf][v] + bn * corrv[m];
      }
    }
  }
}

extern "C" void kernel_launch(void* const* d_in, const int* in_sizes, int n_in,
                              void* d_out, int out_size, void* d_ws, size_t ws_size,
                              hipStream_t stream) {
  const float* x      = (const float*)d_in[0];
  const float* weight = (const float*)d_in[1];
  const float* bias   = (const float*)d_in[2];
  float* outp = (float*)d_out;

  // ws layout: [0..1] minmax keys | pad | geff_sum[1024] | gq_sum[1024]
  //            | corr[8192] | Weff f16 [1024*1024]   (~2.1 MB total)
  unsigned* keys   = (unsigned*)d_ws;
  float* geff_sum  = (float*)d_ws + 16;
  float* gq_sum    = geff_sum + IN_F;
  float* corrv     = gq_sum + IN_F;
  _Float16* Weff   = (_Float16*)(corrv + BATCH);

  xbar_init_kernel<<<8, 256, 0, stream>>>(keys, geff_sum);
  xbar_minmax_kernel<<<256, 256, 0, stream>>>(weight, keys);
  xbar_prep_kernel<<<OUT_F / 16, 256, 0, stream>>>(weight, keys, geff_sum, gq_sum, Weff);
  xbar_corr_kernel<<<BATCH / 8, 256, 0, stream>>>(x, geff_sum, gq_sum, corrv);
  dim3 grid(OUT_F / 128, BATCH / 128);
  xbar_gemm_kernel<<<grid, 256, 0, stream>>>(x, Weff, bias, corrv, outp);
}
